// GatedSSM_11802570130104
// MI455X (gfx1250) — matrix operand order repaired and checked by reading, not compiled, not run on hardware
//
#include <hip/hip_runtime.h>
#include <hip/hip_bf16.h>

// ---------------------------------------------------------------------------
// Gated SSM (GRU-like) persistent-kernel implementation for gfx1250 (MI455X).
//
//  * 16 persistent workgroups (1 per WGP), 256 threads (8 wave32 each).
//  * WG g owns output columns [g*64, g*64+64): 64-row slices of W_z / W_h
//    pinned in LDS as bf16 (~258 KB, padded rows -> conflict-free b128 reads).
//  * Each wave owns one 16x16 output tile; K=1024 runs as 32 x
//    v_wmma_f32_16x16x32_bf16 per GEMM, A fragment shared by both GEMMs.
//    fp32 hidden state for the tile stays in registers for all T steps.
//  * Recurrence crosses WGs via double-buffered bf16 h in global (L2) plus
//    one grid barrier per step (monotonic atomic arrive + s_sleep spin).
//  * Gates are branch-free (exp-based sigmoid/tanh + fast reciprocal).
//  * A-stream prefetch is hoisted OUT of the K loop (straight-line
//    global_prefetch_b8 batch per step) so the K loop is pure load+WMMA.
// ---------------------------------------------------------------------------

typedef __attribute__((ext_vector_type(16))) __bf16    bf16x16;
typedef __attribute__((ext_vector_type(8)))  float     f32x8;
typedef __attribute__((ext_vector_type(4)))  unsigned  u32x4;

union Frag {
    u32x4   u[2];
    bf16x16 v;
};

constexpr int   DD      = 1024;   // hidden dim
constexpr int   BB      = 32;     // batch
constexpr int   TT      = 1024;   // sequence length
constexpr int   NWG     = 16;     // persistent workgroups
constexpr int   NSLICE  = 64;     // output columns per WG
constexpr int   LSTR    = 1032;   // LDS row stride in bf16 elems (1024 + 8 pad)
constexpr float EPS     = 1e-7f;

// workspace layout (bytes)
constexpr size_t WS_HBUF   = 0;                                  // bf16[2][BB][DD]
constexpr size_t WS_YACC   = 2ull * BB * DD * sizeof(__bf16);    // float[TT][BB]
constexpr size_t WS_ARRIVE = WS_YACC + (size_t)TT * BB * sizeof(float);

// branch-free fast sigmoid: 1/(1+exp(-a))  -> v_exp_f32 + v_rcp_f32
__device__ __forceinline__ float fast_sigmoid(float a) {
    return __fdividef(1.0f, 1.0f + __expf(-a));
}
// branch-free fast tanh: copysign((1-e)/(1+e), a), e = exp(-2|a|)
__device__ __forceinline__ float fast_tanh(float a) {
    const float e = __expf(-2.0f * __builtin_fabsf(a));
    return __builtin_copysignf(__fdividef(1.0f - e, 1.0f + e), a);
}

__global__ void init_ws_kernel(__bf16* __restrict__ hbuf,
                               float* __restrict__ yacc,
                               unsigned* __restrict__ arrive) {
    int i = blockIdx.x * blockDim.x + threadIdx.x;
    int n = blockDim.x * gridDim.x;
    for (int k = i; k < 2 * BB * DD; k += n) hbuf[k] = (__bf16)0.0f;
    for (int k = i; k < TT * BB; k += n)     yacc[k] = 0.0f;
    if (i == 0) *arrive = 0u;
}

__global__ void __launch_bounds__(256, 1)
ssm_persistent_kernel(const float* __restrict__ x,     // [BB][TT]
                      const float* __restrict__ Wz,    // [DD][DD]
                      const float* __restrict__ Uz,    // [DD]
                      const float* __restrict__ bz,    // [DD]
                      const float* __restrict__ Wh,    // [DD][DD]
                      const float* __restrict__ Uh,    // [DD]
                      const float* __restrict__ bh,    // [DD]
                      const float* __restrict__ Cv,    // [DD]
                      __bf16* __restrict__ hbuf,       // [2][BB][DD]
                      float* __restrict__ yacc,        // [TT][BB]
                      unsigned* __restrict__ arrive) {
    const int g    = blockIdx.x;       // owns cols [g*64, g*64+64)
    const int tid  = threadIdx.x;
    const int lane = tid & 31;
    const int wave = tid >> 5;         // 0..7
    const int mt   = wave >> 2;        // M tile: 0..1
    const int nt   = wave & 3;         // N tile: 0..3
    const int grp  = lane >> 4;        // half-wave: 0/1
    const int l16  = lane & 15;

    extern __shared__ __bf16 smem[];
    __bf16* Wz_l = smem;                        // [64][LSTR] bf16
    __bf16* Wh_l = smem + NSLICE * LSTR;        // [64][LSTR] bf16

    // ---- stage weight slices fp32 -> bf16 into LDS (once) -----------------
    for (int idx = tid; idx < NSLICE * DD; idx += 256) {
        const int r = idx >> 10;
        const int c = idx & (DD - 1);
        Wz_l[r * LSTR + c] = (__bf16)Wz[(size_t)(g * NSLICE + r) * DD + c];
        Wh_l[r * LSTR + c] = (__bf16)Wh[(size_t)(g * NSLICE + r) * DD + c];
    }
    __syncthreads();

    // ---- per-lane constants ----------------------------------------------
    const int ncol   = g * NSLICE + nt * 16 + l16;  // this lane's output column
    const float u_z  = Uz[ncol];
    const float u_h  = Uh[ncol];
    const float bb_z = bz[ncol];
    const float bb_h = bh[ncol];
    const float c_n  = Cv[ncol];

    const int mbase = mt * 16 + grp * 8;  // C/D tile: VGPR v holds row mbase+v
    const int mrowA = mt * 16 + l16;      // A fragment: this lane's batch row

    const __bf16* bzp = Wz_l + (size_t)(nt * 16 + l16) * LSTR + grp * 16;
    const __bf16* bhp = Wh_l + (size_t)(nt * 16 + l16) * LSTR + grp * 16;

    f32x8 hold;
    #pragma unroll
    for (int i = 0; i < 8; ++i) hold[i] = 0.0f;

    for (int t = 0; t < TT; ++t) {
        const __bf16* hrd = hbuf + (size_t)(t & 1) * (BB * DD);
        __bf16*       hwr = hbuf + (size_t)((t + 1) & 1) * (BB * DD);
        const __bf16* ap  = hrd + (size_t)mrowA * DD + grp * 16;

        // warm WGP$ for this step's A stream -- straight-line, outside K loop
        #pragma unroll
        for (int pf = 0; pf < 4; ++pf)
            __builtin_prefetch(ap + pf * 512, 0, 3);   // global_prefetch_b8

        f32x8 accz, acch;
        #pragma unroll
        for (int i = 0; i < 8; ++i) { accz[i] = 0.0f; acch[i] = 0.0f; }

        // ---- K loop: 32 steps of 16x16x32 bf16 WMMA x 2 GEMMs ------------
        #pragma unroll 4
        for (int kk = 0; kk < DD / 32; ++kk) {
            Frag fa, fz, fh;
            const u32x4* gp = (const u32x4*)(ap + kk * 32);
            fa.u[0] = gp[0];
            fa.u[1] = gp[1];
            const u32x4* lz = (const u32x4*)(bzp + kk * 32);
            fz.u[0] = lz[0];
            fz.u[1] = lz[1];
            const u32x4* lh = (const u32x4*)(bhp + kk * 32);
            fh.u[0] = lh[0];
            fh.u[1] = lh[1];

            accz = __builtin_amdgcn_wmma_f32_16x16x32_bf16(
                false, fa.v, false, fz.v, (short)0, accz, false, false);
            acch = __builtin_amdgcn_wmma_f32_16x16x32_bf16(
                false, fa.v, false, fh.v, (short)0, acch, false, false);
        }

        // ---- gates + state update + readout partials (branch-free) -------
        #pragma unroll
        for (int v = 0; v < 8; ++v) {
            const int m  = mbase + v;
            const float xv = x[(size_t)m * TT + t];
            const float az = accz[v] + xv * u_z + bb_z;
            const float ah = acch[v] + xv * u_h + bb_h;
            const float zg = fast_sigmoid(az);
            const float th = fast_tanh(ah);
            const float hn = hold[v] + zg * (th - hold[v]);  // (1-z)h + z*h~
            hold[v] = hn;

            hwr[(size_t)m * DD + ncol] = (__bf16)hn;

            float p = hn * c_n;
            p += __shfl_xor(p, 1, 16);
            p += __shfl_xor(p, 2, 16);
            p += __shfl_xor(p, 4, 16);
            p += __shfl_xor(p, 8, 16);
            if (l16 == 0) atomicAdd(&yacc[t * BB + m], p);
        }

        // ---- grid barrier ------------------------------------------------
        __syncthreads();
        if (tid == 0) {
            __threadfence();            // release h stores + atomics
            atomicAdd(arrive, 1u);
            const unsigned target = (unsigned)(t + 1) * NWG;
            while (__hip_atomic_load(arrive, __ATOMIC_RELAXED,
                                     __HIP_MEMORY_SCOPE_AGENT) < target) {
                __builtin_amdgcn_s_sleep(2);
            }
            __threadfence();            // acquire
        }
        __syncthreads();
    }
}

__global__ void finalize_kernel(const float* __restrict__ yacc,
                                const float* __restrict__ dsc,
                                float* __restrict__ out) {
    const int i = blockIdx.x * blockDim.x + threadIdx.x;  // over BB*TT
    if (i < BB * TT) {
        const int b = i / TT;
        const int t = i % TT;
        float s = fast_sigmoid(yacc[t * BB + b] + dsc[0]);
        s = fminf(fmaxf(s, EPS), 1.0f - EPS);
        out[(size_t)b * TT + t] = s;
    }
}

extern "C" void kernel_launch(void* const* d_in, const int* in_sizes, int n_in,
                              void* d_out, int out_size, void* d_ws, size_t ws_size,
                              hipStream_t stream) {
    (void)in_sizes; (void)n_in; (void)out_size; (void)ws_size;

    const float* x   = (const float*)d_in[0];
    const float* Wz  = (const float*)d_in[1];
    const float* Uz  = (const float*)d_in[2];
    const float* bz  = (const float*)d_in[3];
    const float* Wh  = (const float*)d_in[4];
    const float* Uh  = (const float*)d_in[5];
    const float* bh  = (const float*)d_in[6];
    const float* Cv  = (const float*)d_in[7];
    const float* dsc = (const float*)d_in[8];
    float* out = (float*)d_out;

    char* ws = (char*)d_ws;
    __bf16*   hbuf   = (__bf16*)(ws + WS_HBUF);
    float*    yacc   = (float*)(ws + WS_YACC);
    unsigned* arrive = (unsigned*)(ws + WS_ARRIVE);

    init_ws_kernel<<<dim3(64), dim3(256), 0, stream>>>(hbuf, yacc, arrive);

    const size_t smem_bytes = (size_t)2 * NSLICE * LSTR * sizeof(__bf16); // 264192 < 320K
    ssm_persistent_kernel<<<dim3(NWG), dim3(256), smem_bytes, stream>>>(
        x, Wz, Uz, bz, Wh, Uh, bh, Cv, hbuf, yacc, arrive);

    finalize_kernel<<<dim3((BB * TT + 255) / 256), dim3(256), 0, stream>>>(
        yacc, dsc, out);
}
